// TSNN_54640573939716
// MI455X (gfx1250) — matrix operand order repaired and checked by reading, not compiled, not run on hardware
//
#include <hip/hip_runtime.h>
#include <hip/hip_bf16.h>
#include <math.h>

typedef __attribute__((ext_vector_type(16))) _Float16 v16h;
typedef __attribute__((ext_vector_type(8)))  _Float16 v8h;
typedef __attribute__((ext_vector_type(8)))  float    v8f;
typedef __attribute__((ext_vector_type(4)))  float    f4;

#define NF     256
#define NHIDC  128
#define NCLSC  4
#define GG     64
#define LLEN   512
#define NNODE  (GG*LLEN)          // 32768
#define NEDGE  524288
#define ETOT   (NEDGE+NNODE)
#define NHEADS 2
#define HDIM   128
#define DFFC   128
#define EPSC   1e-5f
#define NEGV   (-1000000000.0f)

// ---------- device helpers ----------
__device__ __forceinline__ int fenc(float f){ int i=__float_as_int(f); return (i>=0)? i : (i ^ 0x7fffffff); }
__device__ __forceinline__ float fdec(int i){ return __int_as_float((i>=0)? i : (i ^ 0x7fffffff)); }

__device__ __forceinline__ float wred_add(float v){
  #pragma unroll
  for (int o=16;o;o>>=1) v += __shfl_xor(v,o,32);
  return v;
}
__device__ __forceinline__ float wred_max(float v){
  #pragma unroll
  for (int o=16;o;o>>=1) v = fmaxf(v, __shfl_xor(v,o,32));
  return v;
}

// CDNA5 async global->LDS DMA: one b128 chunk per lane, tracked by ASYNCcnt.
// lds_off: byte offset of destination in LDS; gptr: per-lane global source.
__device__ __forceinline__ void async_g2l_b128(unsigned lds_off, const float* gptr)
{
  unsigned long long ga = (unsigned long long)(uintptr_t)gptr;
  asm volatile("global_load_async_to_lds_b128 %0, %1, off"
               :: "v"(lds_off), "v"(ga)
               : "memory");
}
__device__ __forceinline__ void wait_async0()
{
#if __has_builtin(__builtin_amdgcn_s_wait_asynccnt)
  __builtin_amdgcn_s_wait_asynccnt(0);
#else
  asm volatile("s_wait_asynccnt 0x0" ::: "memory");
#endif
}

// convert 8 contiguous f32 in LDS (16B-aligned) into fragment halves [base..base+7]
__device__ __forceinline__ void cvt8(const float* __restrict__ p, v16h& r, int base)
{
  f4 a = *(const f4*)p;
  f4 b = *(const f4*)(p + 4);
  #pragma unroll
  for (int i = 0; i < 4; ++i) {
    r[base + i]     = (_Float16)a[i];
    r[base + 4 + i] = (_Float16)b[i];
  }
}

// ---------- generic batched WMMA GEMM ----------
// C[m,n] = act( alpha * sum_k A[m,k]*B(k,n) + bias[n] )
// opB==0 : B(k,n) = B[k*ldb+n] (row-major [K,N])
// opB==1 : B(k,n) = B[n*ldb+k] (row-major [N,K], i.e. x @ W^T)
// batch z: g=z/nh, h=z%nh, operand base offsets g*?gs + h*?hs
// Requires: N % 64 == 0, K % 32 == 0. M arbitrary (A row addresses clamped,
// stores bounds-guarded).
__global__ __launch_bounds__(256)
void k_wmma_gemm(const float* __restrict__ A, const float* __restrict__ B,
                 const float* __restrict__ bias, float* __restrict__ C,
                 int M, int N, int K, int lda, int ldb, int ldc,
                 int opB, int act, float alpha, int nh,
                 long long ags, long long ahs, long long bgs, long long bhs,
                 long long cgs, long long chs)
{
  const int z  = blockIdx.z;
  const int gi = z / nh, hi = z - gi*nh;
  A += gi*ags + hi*ahs;
  B += gi*bgs + hi*bhs;
  C += gi*cgs + hi*chs;

  constexpr int BM = 128, BN = 64, BK = 32;
  __shared__ float As32[BM*BK];   // [m][k]                          (16 KB)
  __shared__ float Bs32[BN*BK];   // opB=1: [n][k] ; opB=0: [k][n]   (8 KB)

  const int tid  = threadIdx.x;
  const int wave = tid >> 5;
  const int lane = tid & 31;
  const int wm   = (wave & 3) * 32;
  const int wn   = (wave >> 2) * 32;
  const int bm0  = blockIdx.y * BM;
  const int bn0  = blockIdx.x * BN;

  const unsigned asBase = (unsigned)(uintptr_t)(void*)As32;
  const unsigned bsBase = (unsigned)(uintptr_t)(void*)Bs32;

  v8f acc[2][2] = {};

  const int lm  = lane & 15;
  const int ak0 = (lane < 16) ? 0 : 16;  // A frag: lanes 16-31 take K=16..31
  const int bk0 = (lane < 16) ? 0 : 16;  // B frag: lanes 16-31 take K=16..31

  for (int kt = 0; kt < K; kt += BK) {
    // ---- async-DMA stage A tile: 128x32 f32 = 1024 x b128 chunks ----
    #pragma unroll
    for (int c = tid; c < BM*BK/4; c += 256) {
      int m    = c >> 3;
      int koff = (c & 7) << 2;
      int gm   = bm0 + m;
      gm = (gm < M) ? gm : (M - 1);   // clamp: keeps address valid, rows unused
      async_g2l_b128(asBase + c*16, A + (long long)gm*lda + (kt + koff));
    }
    // ---- async-DMA stage B tile: 64x32 f32 = 512 x b128 chunks ----
    if (opB) {
      #pragma unroll
      for (int c = tid; c < BN*BK/4; c += 256) {
        int n    = c >> 3;
        int koff = (c & 7) << 2;
        async_g2l_b128(bsBase + c*16, B + (long long)(bn0 + n)*ldb + (kt + koff));
      }
    } else {
      #pragma unroll
      for (int c = tid; c < BN*BK/4; c += 256) {
        int k    = c >> 4;
        int noff = (c & 15) << 2;
        async_g2l_b128(bsBase + c*16, B + (long long)(kt + k)*ldb + (bn0 + noff));
      }
    }
    wait_async0();
    __syncthreads();

    // ---- fragment reads (ds_load_b128) + f32->f16 packing ----
    v16h af[2], bf[2];
    #pragma unroll
    for (int r = 0; r < 2; ++r) {
      const float* ap = &As32[(wm + r*16 + lm)*BK];
      cvt8(ap + ak0,      af[r], 0);
      cvt8(ap + ak0 + 8,  af[r], 8);
    }
    if (opB) {
      #pragma unroll
      for (int r = 0; r < 2; ++r) {
        const float* bp = &Bs32[(wn + r*16 + lm)*BK + bk0];
        cvt8(bp,     bf[r], 0);
        cvt8(bp + 8, bf[r], 8);
      }
    } else {
      #pragma unroll
      for (int r = 0; r < 2; ++r) {
        int n = wn + r*16 + lm;
        v16h b;
        #pragma unroll
        for (int e = 0; e < 16; ++e) b[e] = (_Float16)Bs32[(bk0 + e)*BN + n];
        bf[r] = b;
      }
    }

    #pragma unroll
    for (int r = 0; r < 2; ++r)
      #pragma unroll
      for (int c = 0; c < 2; ++c)
        acc[r][c] = __builtin_amdgcn_wmma_f32_16x16x32_f16(
            false, af[r], false, bf[c], (short)0, acc[r][c], false, false);
    __syncthreads();
  }

  const int mo = (lane < 16) ? 0 : 8;
  #pragma unroll
  for (int r = 0; r < 2; ++r) {
    #pragma unroll
    for (int c = 0; c < 2; ++c) {
      int n = bn0 + wn + c*16 + lm;
      float bv = bias ? bias[n] : 0.f;
      #pragma unroll
      for (int v = 0; v < 8; ++v) {
        int m = bm0 + wm + r*16 + mo + v;
        if (m < M) {
          float val = acc[r][c][v] * alpha + bv;
          if (act) val = fmaxf(val, 0.f);
          C[(long long)m*ldc + n] = val;
        }
      }
    }
  }
}

// ---------- graph kernels ----------
__global__ void k_fill_mx(int* mx, int n){
  int i = blockIdx.x*blockDim.x + threadIdx.x;
  if (i < n) mx[i] = fenc(-INFINITY);
}

__global__ __launch_bounds__(256)
void k_rowdot2(const float* __restrict__ h, const float* __restrict__ w1,
               const float* __restrict__ w2, float* __restrict__ o1,
               float* __restrict__ o2, int rows)
{
  int r = blockIdx.x*8 + (threadIdx.x>>5); if (r>=rows) return;
  int lane = threadIdx.x & 31;
  const float* hr = h + (size_t)r*NF;
  float s1=0.f, s2=0.f;
  #pragma unroll
  for (int j=0;j<NF/32;++j){ int f=lane+j*32; float x=hr[f]; s1+=x*w1[f]; s2+=x*w2[f]; }
  s1 = wred_add(s1); s2 = wred_add(s2);
  if (lane==0){ o1[r]=s1; o2[r]=s2; }
}

__global__ __launch_bounds__(256)
void k_padmask(const float* __restrict__ x, float* __restrict__ pad, int rows)
{
  int r = blockIdx.x*8 + (threadIdx.x>>5); if (r>=rows) return;
  int lane = threadIdx.x & 31;
  const float* xr = x + (size_t)r*NF;
  float s=0.f;
  #pragma unroll
  for (int j=0;j<NF/32;++j) s += xr[lane+j*32];
  s = wred_add(s);
  if (lane==0) pad[r] = (s==0.f) ? 1.f : 0.f;
}

__global__ void k_edge_logit(const int* __restrict__ ei, const float* __restrict__ a_s,
                             const float* __restrict__ a_d, float* __restrict__ logit,
                             int* __restrict__ mx)
{
  int e = blockIdx.x*blockDim.x + threadIdx.x; if (e >= ETOT) return;
  int s, d; bool keep;
  if (e < NEDGE){ s = ei[e]; d = ei[NEDGE+e]; keep = (s % LLEN != 0) && (d % LLEN != 0); }
  else          { s = d = e - NEDGE; keep = true; }
  float lg = a_s[s] + a_d[d];
  lg = (lg > 0.f) ? lg : 0.2f*lg;           // leaky_relu slope 0.2
  if (!keep) lg = NEGV;
  logit[e] = lg;
  atomicMax(&mx[d], fenc(lg));
}

__global__ void k_edge_exp(const int* __restrict__ ei, const float* __restrict__ logit,
                           const int* __restrict__ mx, float* __restrict__ ex,
                           float* __restrict__ den)
{
  int e = blockIdx.x*blockDim.x + threadIdx.x; if (e >= ETOT) return;
  int d = (e < NEDGE) ? ei[NEDGE+e] : (e - NEDGE);
  float v = __expf(logit[e] - fdec(mx[d]));
  ex[e] = v;
  atomicAdd(&den[d], v);
}

__global__ __launch_bounds__(256)
void k_gat_agg(const int* __restrict__ ei, const float* __restrict__ ex,
               const float* __restrict__ den, const float* __restrict__ h,
               float* __restrict__ xg)
{
  int e = blockIdx.x*8 + (threadIdx.x>>5); if (e >= ETOT) return;
  int lane = threadIdx.x & 31;
  int s, d;
  if (e < NEDGE){ s = ei[e]; d = ei[NEDGE+e]; } else { s = d = e - NEDGE; }
  float alpha = ex[e] / den[d];
  const float* hs = h + (size_t)s*NF;
  float* xd = xg + (size_t)d*NF;
  #pragma unroll
  for (int j=0;j<NF/32;++j){ int f=lane+j*32; atomicAdd(&xd[f], alpha*hs[f]); }
}

__global__ void k_wf_deg(const int* __restrict__ ei, const float* __restrict__ tds,
                         const float* __restrict__ dep, float* __restrict__ wf,
                         float* __restrict__ deg)
{
  int e = blockIdx.x*blockDim.x + threadIdx.x; if (e >= ETOT) return;
  int d; float w;
  if (e < NEDGE){ d = ei[NEDGE+e]; w = tds[e]/dep[e]; } else { d = e-NEDGE; w = 1.f; }
  wf[e] = w;
  atomicAdd(&deg[d], w);
}

__global__ void k_dinv(const float* __restrict__ deg, float* __restrict__ dinv, int n)
{
  int i = blockIdx.x*blockDim.x + threadIdx.x; if (i>=n) return;
  float dg = deg[i];
  dinv[i] = (dg > 0.f) ? rsqrtf(dg) : 0.f;
}

__global__ void k_norm(const int* __restrict__ ei, const float* __restrict__ wf,
                       const float* __restrict__ dinv, float* __restrict__ nrm)
{
  int e = blockIdx.x*blockDim.x + threadIdx.x; if (e >= ETOT) return;
  int s, d;
  if (e < NEDGE){ s = ei[e]; d = ei[NEDGE+e]; } else { s = d = e - NEDGE; }
  nrm[e] = dinv[s]*wf[e]*dinv[d];
}

__global__ __launch_bounds__(256)
void k_gcn_agg(const int* __restrict__ ei, const float* __restrict__ nrm,
               const float* __restrict__ hh, float* __restrict__ out, int Fo)
{
  int e = blockIdx.x*8 + (threadIdx.x>>5); if (e >= ETOT) return;
  int lane = threadIdx.x & 31;
  int s, d;
  if (e < NEDGE){ s = ei[e]; d = ei[NEDGE+e]; } else { s = d = e - NEDGE; }
  float w = nrm[e];
  const float* hs = hh + (size_t)s*Fo;
  float* od = out + (size_t)d*Fo;
  for (int j=0;j<Fo/32;++j){ int f=lane+j*32; atomicAdd(&od[f], w*hs[f]); }
}

__global__ void k_bias_relu(float* __restrict__ io, const float* __restrict__ bias,
                            long long total, int Fo)
{
  long long i = (long long)blockIdx.x*blockDim.x + threadIdx.x;
  if (i >= total) return;
  int f = (int)(i % Fo);
  io[i] = fmaxf(io[i] + bias[f], 0.f);
}

// ---------- transformer helpers ----------
__global__ void k_make_s(const float* __restrict__ x, float* __restrict__ s, long long total)
{
  long long i = (long long)blockIdx.x*blockDim.x + threadIdx.x;
  if (i >= total) return;
  int f = (int)(i % NF);
  int l = (int)((i / NF) % LLEN);
  int f2 = (f >> 1) << 1;
  float div = __expf(-(float)f2 * (9.210340371976184f / 256.0f));
  float ang = (float)l * div;
  float pe  = (f & 1) ? __cosf(ang) : __sinf(ang);
  s[i] = x[i]*16.0f + pe;
}

__global__ __launch_bounds__(256)
void k_softmax(float* __restrict__ sc, const float* __restrict__ pad, int useMask, int rows)
{
  int r = blockIdx.x*8 + (threadIdx.x>>5); if (r>=rows) return;
  int lane = threadIdx.x & 31;
  int z = r / LLEN;
  int g = z / NHEADS;
  float* row = sc + (size_t)r*LLEN;
  const float* pm = pad + (size_t)g*LLEN;
  float v[LLEN/32];
  float mx = -INFINITY;
  #pragma unroll
  for (int j=0;j<LLEN/32;++j){
    int kk = lane + j*32;
    float val = row[kk];
    if (useMask && pm[kk] != 0.f) val = NEGV;
    v[j] = val; mx = fmaxf(mx, val);
  }
  mx = wred_max(mx);
  float sum = 0.f;
  #pragma unroll
  for (int j=0;j<LLEN/32;++j){ float e = __expf(v[j]-mx); v[j]=e; sum += e; }
  sum = wred_add(sum);
  float inv = 1.f/sum;
  #pragma unroll
  for (int j=0;j<LLEN/32;++j) row[lane + j*32] = v[j]*inv;
}

// in-place LayerNorm with optional residual: io = LN(io + resid) * g + b
__global__ __launch_bounds__(256)
void k_add_ln(float* __restrict__ io, const float* __restrict__ resid,
              const float* __restrict__ gw, const float* __restrict__ bw, int rows)
{
  int r = blockIdx.x*8 + (threadIdx.x>>5); if (r>=rows) return;
  int lane = threadIdx.x & 31;
  float* row = io + (size_t)r*NF;
  const float* rr = resid ? resid + (size_t)r*NF : nullptr;
  float v[NF/32];
  float s=0.f, s2=0.f;
  #pragma unroll
  for (int j=0;j<NF/32;++j){
    int f = lane + j*32;
    float x = row[f] + (rr ? rr[f] : 0.f);
    v[j] = x; s += x; s2 += x*x;
  }
  s = wred_add(s); s2 = wred_add(s2);
  float mu = s * (1.0f/NF);
  float var = s2 * (1.0f/NF) - mu*mu;
  float rs = rsqrtf(var + EPSC);
  #pragma unroll
  for (int j=0;j<NF/32;++j){
    int f = lane + j*32;
    row[f] = (v[j]-mu)*rs*gw[f] + bw[f];
  }
}

// ---------- head kernels ----------
__global__ void k_gather_last(const float* __restrict__ t, float* __restrict__ out)
{
  int i = blockIdx.x*blockDim.x + threadIdx.x;
  if (i >= GG*NF) return;
  int g = i / NF, f = i % NF;
  out[i] = t[((size_t)g*LLEN + (LLEN-1))*NF + f];
}

__global__ void k_concat_cls(const float* __restrict__ xgc, const float* __restrict__ Sb,
                             float* __restrict__ cls)
{
  int i = blockIdx.x*blockDim.x + threadIdx.x;
  if (i >= GG*NF) return;
  int g = i / NF, f = i % NF;
  cls[i] = (f < NHIDC) ? xgc[(size_t)(g*LLEN)*NHIDC + f]   // news_index[g] = g*L
                       : Sb[(size_t)g*NHIDC + (f - NHIDC)];
}

__global__ __launch_bounds__(256)
void k_cls(const float* __restrict__ feat, const float* __restrict__ W,
           const float* __restrict__ b, float* __restrict__ out)
{
  int g = blockIdx.x*8 + (threadIdx.x>>5); if (g>=GG) return;
  int lane = threadIdx.x & 31;
  const float* fr = feat + (size_t)g*NF;
  float a0=0,a1=0,a2=0,a3=0;
  for (int f=lane; f<NF; f+=32){
    float x = fr[f];
    const float* w = W + (size_t)f*NCLSC;
    a0 += x*w[0]; a1 += x*w[1]; a2 += x*w[2]; a3 += x*w[3];
  }
  a0=wred_add(a0); a1=wred_add(a1); a2=wred_add(a2); a3=wred_add(a3);
  if (lane==0){
    float l0=a0+b[0], l1=a1+b[1], l2=a2+b[2], l3=a3+b[3];
    float m = fmaxf(fmaxf(l0,l1), fmaxf(l2,l3));
    float se = __expf(l0-m)+__expf(l1-m)+__expf(l2-m)+__expf(l3-m);
    float ls = m + __logf(se);
    out[g*4+0]=l0-ls; out[g*4+1]=l1-ls; out[g*4+2]=l2-ls; out[g*4+3]=l3-ls;
  }
}

// ---------- host side ----------
static inline void gemm(hipStream_t st, const float* A, const float* B, const float* bias,
                        float* C, int M, int N, int K, int lda, int ldb, int ldc,
                        int opB, int act, float alpha,
                        int Z, int nh,
                        long long ags, long long ahs, long long bgs, long long bhs,
                        long long cgs, long long chs)
{
  dim3 grid(N/64, (M+127)/128, Z);
  k_wmma_gemm<<<grid, 256, 0, st>>>(A,B,bias,C,M,N,K,lda,ldb,ldc,opB,act,alpha,
                                    nh,ags,ahs,bgs,bhs,cgs,chs);
}
static inline void gemm1(hipStream_t st, const float* A, const float* B, const float* bias,
                         float* C, int M, int N, int K, int lda, int ldb, int ldc,
                         int opB, int act, float alpha)
{
  gemm(st, A,B,bias,C,M,N,K,lda,ldb,ldc,opB,act,alpha, 1,1, 0,0,0,0,0,0);
}

static void run_mha(hipStream_t st, const float* qin, const float* kvin,
                    const float* inw, const float* inb, const float* ow, const float* ob,
                    const float* pad, float* out,
                    float* q, float* k, float* v, float* sc, float* attn)
{
  const int F = NF;
  // Q/K/V projections (weights are [F,F] row-major, applied transposed)
  gemm1(st, qin,  inw,          inb,        q, NNODE, F, F, F, F, F, 1, 0, 1.f);
  gemm1(st, kvin, inw + F*F,    inb + F,    k, NNODE, F, F, F, F, F, 1, 0, 1.f);
  gemm1(st, kvin, inw + 2*F*F,  inb + 2*F,  v, NNODE, F, F, F, F, F, 1, 0, 1.f);
  // scores = Q K^T / sqrt(HD), batched over (g,h)
  gemm(st, q, k, nullptr, sc, LLEN, LLEN, HDIM, F, F, LLEN, 1, 0, 0.08838834764831845f,
       GG*NHEADS, NHEADS,
       (long long)LLEN*F, HDIM,            // A: q base = g*L*F + h*HD
       (long long)LLEN*F, HDIM,            // B: k base
       (long long)NHEADS*LLEN*LLEN, (long long)LLEN*LLEN);
  k_softmax<<<dim3((GG*NHEADS*LLEN)/8), 256, 0, st>>>(sc, pad, pad!=nullptr, GG*NHEADS*LLEN);
  // context = P V, written head-merged into attn[32768,256]
  gemm(st, sc, v, nullptr, attn, LLEN, HDIM, LLEN, LLEN, F, F, 0, 0, 1.f,
       GG*NHEADS, NHEADS,
       (long long)NHEADS*LLEN*LLEN, (long long)LLEN*LLEN,
       (long long)LLEN*F, HDIM,
       (long long)LLEN*F, HDIM);
  // output projection
  gemm1(st, attn, ow, ob, out, NNODE, F, F, F, F, F, 1, 0, 1.f);
}

extern "C" void kernel_launch(void* const* d_in, const int* in_sizes, int n_in,
                              void* d_out, int out_size, void* d_ws, size_t ws_size,
                              hipStream_t stream)
{
  const float* x     = (const float*)d_in[0];
  const float* tds   = (const float*)d_in[1];
  const float* depth = (const float*)d_in[2];
  const int*   ei    = (const int*)d_in[3];
  const float* Wgat   = (const float*)d_in[6];
  const float* attsrc = (const float*)d_in[7];
  const float* attdst = (const float*)d_in[8];
  const float* b_gat  = (const float*)d_in[9];
  const float* Wc1    = (const float*)d_in[10];
  const float* bc1    = (const float*)d_in[11];
  const float* Wc2    = (const float*)d_in[12];
  const float* bc2    = (const float*)d_in[13];
  const float* esa_inw = (const float*)d_in[14];
  const float* esa_inb = (const float*)d_in[15];
  const float* esa_ow  = (const float*)d_in[16];
  const float* esa_ob  = (const float*)d_in[17];
  const float* dsa_inw = (const float*)d_in[18];
  const float* dsa_inb = (const float*)d_in[19];
  const float* dsa_ow  = (const float*)d_in[20];
  const float* dsa_ob  = (const float*)d_in[21];
  const float* dca_inw = (const float*)d_in[22];
  const float* dca_inb = (const float*)d_in[23];
  const float* dca_ow  = (const float*)d_in[24];
  const float* dca_ob  = (const float*)d_in[25];
  const float* enc_w1 = (const float*)d_in[26];
  const float* enc_b1 = (const float*)d_in[27];
  const float* enc_w2 = (const float*)d_in[28];
  const float* enc_b2 = (const float*)d_in[29];
  const float* dec_w1 = (const float*)d_in[30];
  const float* dec_b1 = (const float*)d_in[31];
  const float* dec_w2 = (const float*)d_in[32];
  const float* dec_b2 = (const float*)d_in[33];
  const float* eln1g = (const float*)d_in[34];
  const float* eln1b = (const float*)d_in[35];
  const float* eln2g = (const float*)d_in[36];
  const float* eln2b = (const float*)d_in[37];
  const float* dln1g = (const float*)d_in[38];
  const float* dln1b = (const float*)d_in[39];
  const float* dln2g = (const float*)d_in[40];
  const float* dln2b = (const float*)d_in[41];
  const float* dln3g = (const float*)d_in[42];
  const float* dln3b = (const float*)d_in[43];
  const float* encng = (const float*)d_in[44];
  const float* encnb = (const float*)d_in[45];
  const float* decng = (const float*)d_in[46];
  const float* decnb = (const float*)d_in[47];
  const float* Wlin = (const float*)d_in[48];
  const float* blin = (const float*)d_in[49];
  const float* Wcls = (const float*)d_in[50];
  const float* bcls = (const float*)d_in[51];
  float* out = (float*)d_out;

  // workspace bump allocator
  char* wsb = (char*)d_ws;
  size_t off = 0;
  auto alloc = [&](size_t nflt) -> float* {
    float* p = (float*)(wsb + off);
    off += nflt * sizeof(float);
    off = (off + 255) & ~(size_t)255;
    return p;
  };

  float* hbuf   = alloc((size_t)NNODE*NF);     // GAT h, reused as GCN hh
  float* a_s    = alloc(NNODE);
  float* a_d    = alloc(NNODE);
  float* logitb = alloc(ETOT);
  float* exb    = alloc(ETOT);
  int*   mxb    = (int*)alloc(NNODE);
  float* den    = alloc(NNODE);
  float* xg_a   = alloc((size_t)NNODE*NF);
  float* wfb    = alloc(ETOT);
  float* degb   = alloc(NNODE);
  float* dinvb  = alloc(NNODE);
  float* normb  = alloc(ETOT);
  float* xg_b   = alloc((size_t)NNODE*NF);
  float* hh2    = alloc((size_t)NNODE*NHIDC);
  float* xg_c   = alloc((size_t)NNODE*NHIDC);
  float* padb   = alloc(NNODE);
  float* sbuf   = alloc((size_t)NNODE*NF);
  float* memb   = alloc((size_t)NNODE*NF);
  float* tbuf   = alloc((size_t)NNODE*NF);
  float* qb     = alloc((size_t)NNODE*NF);
  float* kb     = alloc((size_t)NNODE*NF);
  float* vb     = alloc((size_t)NNODE*NF);
  float* scores = alloc((size_t)GG*NHEADS*LLEN*LLEN);
  float* attnb  = alloc((size_t)NNODE*NF);
  float* tmpA   = alloc((size_t)NNODE*NF);
  float* ff1    = alloc((size_t)NNODE*DFFC);
  float* seqlast= alloc((size_t)GG*NF);
  float* Sbuf   = alloc((size_t)GG*NHIDC);
  float* clsf   = alloc((size_t)GG*NF);
  (void)ws_size; (void)in_sizes; (void)n_in; (void)out_size;

  auto eg  = [](long long n){ return dim3((unsigned)((n + 255) / 256)); };
  auto wg  = [](long long n){ return dim3((unsigned)((n + 7) / 8)); };   // one wave per item

  // ===================== GAT =====================
  gemm1(stream, x, Wgat, nullptr, hbuf, NNODE, NF, NF, NF, NF, NF, 0, 0, 1.f);
  k_rowdot2<<<wg(NNODE), 256, 0, stream>>>(hbuf, attsrc, attdst, a_s, a_d, NNODE);
  k_fill_mx<<<eg(NNODE), 256, 0, stream>>>(mxb, NNODE);
  hipMemsetAsync(den,  0, (size_t)NNODE*sizeof(float), stream);
  hipMemsetAsync(xg_a, 0, (size_t)NNODE*NF*sizeof(float), stream);
  k_edge_logit<<<eg(ETOT), 256, 0, stream>>>(ei, a_s, a_d, logitb, mxb);
  k_edge_exp  <<<eg(ETOT), 256, 0, stream>>>(ei, logitb, mxb, exb, den);
  k_gat_agg   <<<wg(ETOT), 256, 0, stream>>>(ei, exb, den, hbuf, xg_a);
  k_bias_relu <<<eg((long long)NNODE*NF), 256, 0, stream>>>(xg_a, b_gat, (long long)NNODE*NF, NF);

  // ===================== GCN norm =====================
  hipMemsetAsync(degb, 0, (size_t)NNODE*sizeof(float), stream);
  k_wf_deg<<<eg(ETOT), 256, 0, stream>>>(ei, tds, depth, wfb, degb);
  k_dinv  <<<eg(NNODE), 256, 0, stream>>>(degb, dinvb, NNODE);
  k_norm  <<<eg(ETOT), 256, 0, stream>>>(ei, wfb, dinvb, normb);

  // GCN layer 1: relu(segsum(norm * (xg_a @ Wc1)) + bc1)
  gemm1(stream, xg_a, Wc1, nullptr, hbuf, NNODE, NF, NF, NF, NF, NF, 0, 0, 1.f);
  hipMemsetAsync(xg_b, 0, (size_t)NNODE*NF*sizeof(float), stream);
  k_gcn_agg  <<<wg(ETOT), 256, 0, stream>>>(ei, normb, hbuf, xg_b, NF);
  k_bias_relu<<<eg((long long)NNODE*NF), 256, 0, stream>>>(xg_b, bc1, (long long)NNODE*NF, NF);
  // GCN layer 2 -> 128 features
  gemm1(stream, xg_b, Wc2, nullptr, hh2, NNODE, NHIDC, NF, NF, NHIDC, NHIDC, 0, 0, 1.f);
  hipMemsetAsync(xg_c, 0, (size_t)NNODE*NHIDC*sizeof(float), stream);
  k_gcn_agg  <<<wg(ETOT), 256, 0, stream>>>(ei, normb, hh2, xg_c, NHIDC);
  k_bias_relu<<<eg((long long)NNODE*NHIDC), 256, 0, stream>>>(xg_c, bc2, (long long)NNODE*NHIDC, NHIDC);

  // ===================== transformer inputs =====================
  k_make_s <<<eg((long long)NNODE*NF), 256, 0, stream>>>(x, sbuf, (long long)NNODE*NF);
  k_padmask<<<wg(NNODE), 256, 0, stream>>>(x, padb, NNODE);
  hipMemcpyAsync(memb, sbuf, (size_t)NNODE*NF*sizeof(float), hipMemcpyDeviceToDevice, stream);
  hipMemcpyAsync(tbuf, sbuf, (size_t)NNODE*NF*sizeof(float), hipMemcpyDeviceToDevice, stream);

  // ===================== encoder =====================
  for (int i = 0; i < 2; ++i) {
    run_mha(stream, memb, memb,
            esa_inw + (size_t)i*3*NF*NF, esa_inb + (size_t)i*3*NF,
            esa_ow  + (size_t)i*NF*NF,   esa_ob  + (size_t)i*NF,
            padb, tmpA, qb, kb, vb, scores, attnb);
    k_add_ln<<<wg(NNODE), 256, 0, stream>>>(memb, tmpA, eln1g + i*NF, eln1b + i*NF, NNODE);
    gemm1(stream, memb, enc_w1 + (size_t)i*NF*DFFC, enc_b1 + (size_t)i*DFFC,
          ff1, NNODE, DFFC, NF, NF, DFFC, DFFC, 0, 1, 1.f);
    gemm1(stream, ff1, enc_w2 + (size_t)i*DFFC*NF, enc_b2 + (size_t)i*NF,
          tmpA, NNODE, NF, DFFC, DFFC, NF, NF, 0, 0, 1.f);
    k_add_ln<<<wg(NNODE), 256, 0, stream>>>(memb, tmpA, eln2g + i*NF, eln2b + i*NF, NNODE);
  }
  k_add_ln<<<wg(NNODE), 256, 0, stream>>>(memb, nullptr, encng, encnb, NNODE);

  // ===================== decoder =====================
  for (int i = 0; i < 2; ++i) {
    run_mha(stream, tbuf, tbuf,
            dsa_inw + (size_t)i*3*NF*NF, dsa_inb + (size_t)i*3*NF,
            dsa_ow  + (size_t)i*NF*NF,   dsa_ob  + (size_t)i*NF,
            padb, tmpA, qb, kb, vb, scores, attnb);
    k_add_ln<<<wg(NNODE), 256, 0, stream>>>(tbuf, tmpA, dln1g + i*NF, dln1b + i*NF, NNODE);
    run_mha(stream, tbuf, memb,
            dca_inw + (size_t)i*3*NF*NF, dca_inb + (size_t)i*3*NF,
            dca_ow  + (size_t)i*NF*NF,   dca_ob  + (size_t)i*NF,
            nullptr, tmpA, qb, kb, vb, scores, attnb);
    k_add_ln<<<wg(NNODE), 256, 0, stream>>>(tbuf, tmpA, dln2g + i*NF, dln2b + i*NF, NNODE);
    gemm1(stream, tbuf, dec_w1 + (size_t)i*NF*DFFC, dec_b1 + (size_t)i*DFFC,
          ff1, NNODE, DFFC, NF, NF, DFFC, DFFC, 0, 1, 1.f);
    gemm1(stream, ff1, dec_w2 + (size_t)i*DFFC*NF, dec_b2 + (size_t)i*NF,
          tmpA, NNODE, NF, DFFC, DFFC, NF, NF, 0, 0, 1.f);
    k_add_ln<<<wg(NNODE), 256, 0, stream>>>(tbuf, tmpA, dln3g + i*NF, dln3b + i*NF, NNODE);
  }
  k_add_ln<<<wg(NNODE), 256, 0, stream>>>(tbuf, nullptr, decng, decnb, NNODE);

  // ===================== head =====================
  k_gather_last<<<eg(GG*NF), 256, 0, stream>>>(tbuf, seqlast);
  gemm1(stream, seqlast, Wlin, blin, Sbuf, GG, NHIDC, NF, NF, NHIDC, NHIDC, 0, 1, 1.f);
  k_concat_cls<<<eg(GG*NF), 256, 0, stream>>>(xg_c, Sbuf, clsf);
  k_cls<<<dim3((GG+7)/8), 256, 0, stream>>>(clsf, Wcls, bcls, out);
}
